// NCA_grow_laplace_15401752724244
// MI455X (gfx1250) — matrix operand order repaired and checked by reading, not compiled, not run on hardware
//
#include <hip/hip_runtime.h>

typedef __attribute__((ext_vector_type(16))) _Float16 v16h;
typedef __attribute__((ext_vector_type(8)))  float    v8f;

// Workspace layout (requires 32768 + 64MB):
//   [0,     16384) : w1 f16 fragments  [t(8)][s(2)][lane(32)][e(16)]
//   [16384, 20480) : w2 f16 fragments  [s2(4)][lane(32)][e(16)]
//   [20480, 20992) : b1 f32 reordered  [half(2)][t(8)][r(8)]
//   [32768, ...  ) : x_new f32, B*C*H*W
#define WS_W1F  0
#define WS_W2F  16384
#define WS_B1F  20480
#define WS_XNEW 32768

// ---------------- kernel 0: weight prepack into WMMA fragment order ----------
__global__ __launch_bounds__(256) void nca_prepack(const float* __restrict__ w1,
                                                   const float* __restrict__ b1,
                                                   const float* __restrict__ w2,
                                                   char* __restrict__ ws) {
  int i = blockIdx.x * 256 + threadIdx.x;
  _Float16* w1f = (_Float16*)(ws + WS_W1F);
  _Float16* w2f = (_Float16*)(ws + WS_W2F);
  float*    b1f = (float*)(ws + WS_B1F);
  if (i < 8192) {
    // A-fragment of w1 (M=hid tile 16, K=feature 32): contiguous K = 16*half + e
    int e = i & 15, lane = (i >> 4) & 31, fr = i >> 9;
    int t = fr >> 1, s = fr & 1;
    int li = lane & 15, half = lane >> 4;
    int K = 16 * half + e;
    int f = 32 * s + K;          // feature index 0..63
    int hid = 16 * t + li;       // hidden row 0..127
    w1f[i] = (_Float16)w1[hid * 64 + f];
  } else if (i < 8192 + 2048) {
    // B-fragment of w2^T (K=hid 32, N=outch 16): table order K = 8*half + e (e<8), 16 + 8*half + (e-8) (e>=8)
    int j = i - 8192;
    int e = j & 15, lane = (j >> 4) & 31, s2 = j >> 9;
    int li = lane & 15, half = lane >> 4;
    int hid = 32 * s2 + (e < 8 ? 8 * half + e : 8 + 8 * half + e);
    w2f[j] = (_Float16)w2[li * 128 + hid];
  } else if (i < 8192 + 2048 + 128) {
    int k = i - 8192 - 2048;
    int r = k & 7, t = (k >> 3) & 7, half = k >> 6;
    b1f[k] = b1[16 * t + r + 8 * half];
  }
}

// ---------------- kernel 1: perceive + MLP (WMMA) -> x_new -------------------
__global__ __launch_bounds__(256) void nca_main(const float* __restrict__ x,
                                                const float* __restrict__ rmask,
                                                char* __restrict__ ws) {
  __shared__ float lds[8 * 272];                 // 16x17 padded tile per wave
  int tid  = threadIdx.x;
  int lane = tid & 31, wave = tid >> 5;
  int li   = lane & 15, half = lane >> 4;
  int tile = blockIdx.x * 8 + wave;              // 65536 tiles total
  int w0   = (tile & 15) << 4;
  int hr   = (tile >> 4) & 255;
  int b    = tile >> 12;
  int wcol = w0 + li;                            // this lane's pixel column

  // ---- perceive: build y^T B-fragments (lane = pixel N, 16 consecutive feats)
  v16h yb0{}, yb1{};
#pragma unroll
  for (int s = 0; s < 2; ++s) {
    int cbase = 8 * s + 4 * half;
#pragma unroll
    for (int cc = 0; cc < 4; ++cc) {
      int c = cbase + cc;
      const float* xc = x + ((size_t)(b * 16 + c) << 16) + hr * 256 + wcol;
      float n[3][3];
#pragma unroll
      for (int dh = -1; dh <= 1; ++dh)
#pragma unroll
        for (int dw = -1; dw <= 1; ++dw) {
          bool ok = ((unsigned)(hr + dh) < 256u) && ((unsigned)(wcol + dw) < 256u);
          n[dh + 1][dw + 1] = ok ? xc[dh * 256 + dw] : 0.f;  // zero pad
        }
      float ident = n[1][1];
      float sx = ((n[0][2] - n[0][0]) + 2.f * (n[1][2] - n[1][0]) + (n[2][2] - n[2][0])) * 0.125f;
      float sy = ((n[2][0] - n[0][0]) + 2.f * (n[2][1] - n[0][1]) + (n[2][2] - n[0][2])) * 0.125f;
      float lap = n[0][1] + n[1][0] + n[1][2] + n[2][1] - 4.f * n[1][1];
      v16h& yb = s ? yb1 : yb0;
      yb[4 * cc + 0] = (_Float16)ident;
      yb[4 * cc + 1] = (_Float16)sx;
      yb[4 * cc + 2] = (_Float16)sy;
      yb[4 * cc + 3] = (_Float16)lap;
    }
  }

  // ---- GEMM1: h^T = w1 (A) x y^T (B), K=64 in 2 steps, 8 hid tiles
  const v16h* w1v = (const v16h*)(ws + WS_W1F);
  v8f acc[8];
#pragma unroll
  for (int t = 0; t < 8; ++t) {
    v16h a0 = w1v[(t * 2 + 0) * 32 + lane];
    v16h a1 = w1v[(t * 2 + 1) * 32 + lane];
    v8f z{};
    acc[t] = __builtin_amdgcn_wmma_f32_16x16x32_f16(false, a0, false, yb0, (short)0, z, false, false);
    acc[t] = __builtin_amdgcn_wmma_f32_16x16x32_f16(false, a1, false, yb1, (short)0, acc[t], false, false);
  }

  // ---- bias + relu + pack: D(h^T) fragments are exactly GEMM2 A fragments
  const v8f* b1v = (const v8f*)(ws + WS_B1F);
  v16h a2[4];
#pragma unroll
  for (int s2 = 0; s2 < 4; ++s2) {
    v8f bb0 = b1v[half * 8 + 2 * s2];
    v8f bb1 = b1v[half * 8 + 2 * s2 + 1];
#pragma unroll
    for (int r = 0; r < 8; ++r) {
      a2[s2][r]     = (_Float16)fmaxf(acc[2 * s2][r]     + bb0[r], 0.f);
      a2[s2][8 + r] = (_Float16)fmaxf(acc[2 * s2 + 1][r] + bb1[r], 0.f);
    }
  }

  // ---- GEMM2: dx = h (A) x w2^T (B), K=128 in 4 steps
  const v16h* w2v = (const v16h*)(ws + WS_W2F);
  v8f dacc{};
#pragma unroll
  for (int s2 = 0; s2 < 4; ++s2) {
    v16h b2 = w2v[s2 * 32 + lane];
    dacc = __builtin_amdgcn_wmma_f32_16x16x32_f16(false, a2[s2], false, b2, (short)0, dacc, false, false);
  }

  // ---- epilogue: transpose dx through LDS for coalesced stores
  float* L = lds + wave * 272;
#pragma unroll
  for (int r = 0; r < 8; ++r)
    L[li * 17 + r + 8 * half] = dacc[r];        // [chan=li][pixel=r+8*half]
  __syncthreads();

  float fire = (rmask[((size_t)b << 16) + hr * 256 + wcol] <= 0.5f) ? 1.f : 0.f;
  float* xnew = (float*)(ws + WS_XNEW);
#pragma unroll
  for (int j = 0; j < 8; ++j) {
    int c = 8 * half + j;
    size_t idx = ((size_t)(b * 16 + c) << 16) + hr * 256 + wcol;
    float dxv = L[c * 17 + li];
    xnew[idx] = x[idx] + dxv * fire;
  }
}

// ---------------- kernel 2: life masks + final multiply ----------------------
__global__ __launch_bounds__(256) void nca_finalize(const float* __restrict__ x,
                                                    const char* __restrict__ ws,
                                                    float* __restrict__ out) {
  int idx = blockIdx.x * 256 + threadIdx.x;      // over B*H*W = 1<<20
  int b = idx >> 16, rem = idx & 65535;
  int hr = rem >> 8, wc = rem & 255;
  const float* xnew = (const float*)(ws + WS_XNEW);
  const float* xa = x    + (((size_t)(b * 16 + 3)) << 16);
  const float* na = xnew + (((size_t)(b * 16 + 3)) << 16);
  float pre = -1e30f, post = -1e30f;
#pragma unroll
  for (int dh = -1; dh <= 1; ++dh)
#pragma unroll
    for (int dw = -1; dw <= 1; ++dw) {
      if (((unsigned)(hr + dh) < 256u) && ((unsigned)(wc + dw) < 256u)) {
        int o = (hr + dh) * 256 + wc + dw;
        pre  = fmaxf(pre,  xa[o]);
        post = fmaxf(post, na[o]);
      }
    }
  float life = (pre > 0.1f && post > 0.1f) ? 1.f : 0.f;
#pragma unroll
  for (int c = 0; c < 16; ++c) {
    size_t o = (((size_t)(b * 16 + c)) << 16) + hr * 256 + wc;
    out[o] = xnew[o] * life;
  }
}

extern "C" void kernel_launch(void* const* d_in, const int* in_sizes, int n_in,
                              void* d_out, int out_size, void* d_ws, size_t ws_size,
                              hipStream_t stream) {
  const float* x     = (const float*)d_in[0];
  const float* rmask = (const float*)d_in[1];
  const float* w1    = (const float*)d_in[2];
  const float* b1    = (const float*)d_in[3];
  const float* w2    = (const float*)d_in[4];
  float* out = (float*)d_out;
  char*  ws  = (char*)d_ws;   // needs 32768 + 16*16*256*256*4 bytes

  nca_prepack <<<41,   256, 0, stream>>>(w1, b1, w2, ws);
  nca_main    <<<8192, 256, 0, stream>>>(x, rmask, ws);
  nca_finalize<<<4096, 256, 0, stream>>>(x, ws, out);
}
